// MHA_81793357185327
// MI455X (gfx1250) — compile-verified, matrix operand order checked by reading, not run on hardware
//
#include <hip/hip_runtime.h>
#include <hip/hip_bf16.h>
#include <stdint.h>

// ---------------------------------------------------------------------------
// MHA forward for MI455X (gfx1250, wave32, WMMA 16x16x32 bf16)
//   qkv  = x @ w_qkv^T            (scale folded into q-rows of w_qkv)
//   attn = flash-attention(q,k,v, padding_mask)
//   out  = attn @ w_o^T + b_o
// Global->LDS staging uses CDNA5 async load-to-LDS (ASYNCcnt) when the
// toolchain exposes the builtins; falls back to load+ds_store otherwise.
// ---------------------------------------------------------------------------

typedef __attribute__((ext_vector_type(16))) __bf16 v16bf;
typedef __attribute__((ext_vector_type(8)))  float  v8f;

typedef unsigned short u16;
typedef unsigned int   u32;

#if defined(__has_builtin)
#  if __has_builtin(__builtin_amdgcn_global_load_async_to_lds_b128) && \
      __has_builtin(__builtin_amdgcn_s_wait_asynccnt)
#    define USE_ASYNC_LDS 1
#  endif
#endif
#ifndef USE_ASYNC_LDS
#  define USE_ASYNC_LDS 0
#endif

#if USE_ASYNC_LDS
// Builtin signature (from clang diagnostic): the b128 variant takes pointers
// to a 16-byte int vector, global operand in AS1, LDS operand in AS3.
typedef int v4i_vs __attribute__((vector_size(16)));

// imm offset applies to BOTH the global and LDS address (ISA §15.18 async):
// our tiles advance global and LDS by the same byte delta, so this is exact.
template <int OFF_BYTES>
__device__ __forceinline__ void async_g2l_b128(const void* g, void* l) {
  __builtin_amdgcn_global_load_async_to_lds_b128(
      (__attribute__((address_space(1))) v4i_vs*)(void*)g,
      (__attribute__((address_space(3))) v4i_vs*)l,
      OFF_BYTES, 0);
}
#endif

union __align__(16) Frag {
  v16bf v;
  uint4 u[2];
};

__device__ __forceinline__ u16 f32_to_bf16(float f) {
  union { float f; u32 u; } c;
  c.f = f;
  u32 u = c.u;
  return (u16)((u + 0x7FFFu + ((u >> 16) & 1u)) >> 16);  // RNE
}

__device__ __forceinline__ v8f wmma_bf16(const Frag& a, const Frag& b, v8f c) {
  // (neg_a, A, neg_b, B, c_mod, C, reuse_a, reuse_b)
  return __builtin_amdgcn_wmma_f32_16x16x32_bf16(false, a.v, false, b.v,
                                                 (short)0, c, false, false);
}

// ---------------------------------------------------------------------------
// fp32 -> bf16 conversion (4-wide); elements with flat index < bound get *scale
// ---------------------------------------------------------------------------
__global__ void cvt_f32_bf16_x4(const float4* __restrict__ in,
                                u16* __restrict__ out,
                                int n4, int bound, float scale) {
  int i = blockIdx.x * blockDim.x + threadIdx.x;
  if (i >= n4) return;
  float4 v = in[i];
  float s = ((i * 4) < bound) ? scale : 1.0f;
  ushort4 o;
  o.x = f32_to_bf16(v.x * s);
  o.y = f32_to_bf16(v.y * s);
  o.z = f32_to_bf16(v.z * s);
  o.w = f32_to_bf16(v.w * s);
  *(ushort4*)(out + (size_t)i * 4) = o;
}

// ---------------------------------------------------------------------------
// C[M,N] = A[M,K] * B[N,K]^T (+ bias)   (A,B bf16 row-major, K-contiguous)
// 128x128x32 block tile, 256 threads = 8 waves (4 in M x 2 in N),
// each wave: 32x64 tile = 2x4 WMMA accumulators.
// Async path: double-buffered LDS, DMA of tile k+1 overlaps WMMAs of tile k.
// ---------------------------------------------------------------------------
template <bool OUT_F32, bool HAS_BIAS>
__global__ __launch_bounds__(256)
void gemm_bf16_nt(const u16* __restrict__ A, const u16* __restrict__ B,
                  void* __restrict__ Cv, const float* __restrict__ bias,
                  int M, int N, int K) {
  const int tid  = threadIdx.x;
  const int lane = tid & 31;
  const int wave = tid >> 5;
  const int wm   = wave & 3;   // wave row  (0..3) -> 32 rows each
  const int wn   = wave >> 2;  // wave col  (0..1) -> 64 cols each
  const int lrow = lane & 15;
  const int lhi  = lane >> 4;  // 0/1: selects which 8-wide K-run

  const int bm0 = blockIdx.x * 128;
  const int bn0 = blockIdx.y * 128;

  // cooperative tile load: 2 threads per row, 16 halves (2x b128) each
  const int ldr = tid >> 1;
  const int ldk = (tid & 1) * 16;

  v8f acc[2][4];
#pragma unroll
  for (int i = 0; i < 2; ++i)
#pragma unroll
    for (int j = 0; j < 4; ++j) acc[i][j] = {};

  const u16* gA = A + (size_t)(bm0 + ldr) * K + ldk;
  const u16* gB = B + (size_t)(bn0 + ldr) * K + ldk;

#if USE_ASYNC_LDS
  __shared__ __align__(16) u16 As[2][128][32];  // 16 KB
  __shared__ __align__(16) u16 Bs[2][128][32];  // 16 KB

  // prologue: DMA tile 0 into buffer 0 (4 asyncs / thread, in-order done)
  async_g2l_b128<0>(gA, &As[0][ldr][ldk]);
  async_g2l_b128<16>(gA, &As[0][ldr][ldk]);
  async_g2l_b128<0>(gB, &Bs[0][ldr][ldk]);
  async_g2l_b128<16>(gB, &Bs[0][ldr][ldk]);

  const int nk = K >> 5;
  for (int k = 0; k < nk; ++k) {
    const int p = k & 1;
    if (k + 1 < nk) {
      const u16* nA = gA + (size_t)(k + 1) * 32;
      const u16* nB = gB + (size_t)(k + 1) * 32;
      async_g2l_b128<0>(nA, &As[1 - p][ldr][ldk]);
      async_g2l_b128<16>(nA, &As[1 - p][ldr][ldk]);
      async_g2l_b128<0>(nB, &Bs[1 - p][ldr][ldk]);
      async_g2l_b128<16>(nB, &Bs[1 - p][ldr][ldk]);
      __builtin_amdgcn_s_wait_asynccnt(4);  // tile k landed (in-order)
    } else {
      __builtin_amdgcn_s_wait_asynccnt(0);
    }
    __syncthreads();  // all waves have tile k

    Frag af[2];
#pragma unroll
    for (int i = 0; i < 2; ++i) {
      const int row = wm * 32 + i * 16 + lrow;
      af[i].u[0] = *(const uint4*)&As[p][row][lhi * 8];
      af[i].u[1] = *(const uint4*)&As[p][row][16 + lhi * 8];
    }
    Frag bfr[4];
#pragma unroll
    for (int j = 0; j < 4; ++j) {
      const int col = wn * 64 + j * 16 + lrow;
      bfr[j].u[0] = *(const uint4*)&Bs[p][col][lhi * 8];
      bfr[j].u[1] = *(const uint4*)&Bs[p][col][16 + lhi * 8];
    }
#pragma unroll
    for (int i = 0; i < 2; ++i)
#pragma unroll
      for (int j = 0; j < 4; ++j)
        acc[i][j] = wmma_bf16(af[i], bfr[j], acc[i][j]);

    __syncthreads();  // done reading buf p; safe target for DMA at k+2
  }
#else
  __shared__ __align__(16) u16 As1[128][32];
  __shared__ __align__(16) u16 Bs1[128][32];

  for (int k0 = 0; k0 < K; k0 += 32) {
    uint4 a0 = *(const uint4*)(gA + k0);
    uint4 a1 = *(const uint4*)(gA + k0 + 8);
    uint4 b0 = *(const uint4*)(gB + k0);
    uint4 b1 = *(const uint4*)(gB + k0 + 8);
    __syncthreads();
    *(uint4*)&As1[ldr][ldk]     = a0;
    *(uint4*)&As1[ldr][ldk + 8] = a1;
    *(uint4*)&Bs1[ldr][ldk]     = b0;
    *(uint4*)&Bs1[ldr][ldk + 8] = b1;
    __syncthreads();

    Frag af[2];
#pragma unroll
    for (int i = 0; i < 2; ++i) {
      const int row = wm * 32 + i * 16 + lrow;
      af[i].u[0] = *(const uint4*)&As1[row][lhi * 8];
      af[i].u[1] = *(const uint4*)&As1[row][16 + lhi * 8];
    }
    Frag bfr[4];
#pragma unroll
    for (int j = 0; j < 4; ++j) {
      const int col = wn * 64 + j * 16 + lrow;
      bfr[j].u[0] = *(const uint4*)&Bs1[col][lhi * 8];
      bfr[j].u[1] = *(const uint4*)&Bs1[col][16 + lhi * 8];
    }
#pragma unroll
    for (int i = 0; i < 2; ++i)
#pragma unroll
      for (int j = 0; j < 4; ++j)
        acc[i][j] = wmma_bf16(af[i], bfr[j], acc[i][j]);
  }
#endif

  // epilogue: C layout reg r -> row r + lhi*8, col lrow
#pragma unroll
  for (int i = 0; i < 2; ++i) {
#pragma unroll
    for (int j = 0; j < 4; ++j) {
      const int col = bn0 + wn * 64 + j * 16 + lrow;
      const float bv = HAS_BIAS ? bias[col] : 0.0f;
#pragma unroll
      for (int r = 0; r < 8; ++r) {
        const int row = bm0 + wm * 32 + i * 16 + r + lhi * 8;
        const float v = acc[i][j][r] + bv;
        if (OUT_F32)
          ((float*)Cv)[(size_t)row * N + col] = v;
        else
          ((u16*)Cv)[(size_t)row * N + col] = f32_to_bf16(v);
      }
    }
  }
}

// ---------------------------------------------------------------------------
// Flash attention.  Grid: 4 batches * 16 heads * 16 q-blocks = 1024 blocks.
// Each block: 128 Q rows, 8 waves x 16 rows.  K blocks DMA'd to LDS (async),
// V staged transposed (register path).  Online softmax; P round-trips a
// per-wave LDS tile to convert C-layout -> A-fragment layout for P @ V.
// ---------------------------------------------------------------------------
#define S_LEN 2048
#define QKV_LD 3072

__global__ __launch_bounds__(256)
void attn_flash(const u16* __restrict__ qkv,
                const unsigned char* __restrict__ pmask,
                u16* __restrict__ outb) {
  __shared__ __align__(16) u16 Ks[128][64];      // 16 KB  (key-major)
  __shared__ __align__(16) u16 Vt[64][128];      // 16 KB  (d-major, transposed)
  __shared__ __align__(16) u16 Ps[8][16][128];   // 32 KB  (per-wave P tile)

  const int tid  = threadIdx.x;
  const int lane = tid & 31;
  const int wave = tid >> 5;
  const int lrow = lane & 15;
  const int lhi  = lane >> 4;

  const int idx = blockIdx.x;
  const int qb  = idx & 15;         // q block (128 rows)
  const int h   = (idx >> 4) & 15;  // head
  const int bb  = idx >> 8;         // batch

  const u16* qbase = qkv + (size_t)bb * S_LEN * QKV_LD + h * 64;
  const u16* kbase = qbase + 1024;
  const u16* vbase = qbase + 2048;

  // resident Q fragments: this wave's 16 rows, two 32-wide K slabs over d=64
  const int qrow = qb * 128 + wave * 16 + lrow;
  Frag qf[2];
#pragma unroll
  for (int s = 0; s < 2; ++s) {
    const u16* p = qbase + (size_t)qrow * QKV_LD + s * 32 + lhi * 8;
    qf[s].u[0] = *(const uint4*)p;
    qf[s].u[1] = *(const uint4*)(p + 16);
  }

  float mrun[8], lrun[8];
  v8f o[4];
#pragma unroll
  for (int r = 0; r < 8; ++r) { mrun[r] = -3.0e38f; lrun[r] = 0.0f; }
#pragma unroll
  for (int dt = 0; dt < 4; ++dt) o[dt] = {};

  const unsigned char* mrow = pmask + (size_t)bb * S_LEN;

  for (int kb = 0; kb < 16; ++kb) {
    __syncthreads();  // all waves done reading Ks/Vt from previous block
    {
      // cooperative K/V block load: 2 threads per key, 32 halves each
      const int key = tid >> 1;
      const int dh  = (tid & 1) * 32;
      const u16* kp = kbase + (size_t)(kb * 128 + key) * QKV_LD + dh;
#if USE_ASYNC_LDS
      // K: DMA straight into LDS, overlapped with the V register loads below
      async_g2l_b128<0>(kp, &Ks[key][dh]);
      async_g2l_b128<16>(kp, &Ks[key][dh]);
      async_g2l_b128<32>(kp, &Ks[key][dh]);
      async_g2l_b128<48>(kp, &Ks[key][dh]);
#else
      uint4 k0 = *(const uint4*)(kp);
      uint4 k1 = *(const uint4*)(kp + 8);
      uint4 k2 = *(const uint4*)(kp + 16);
      uint4 k3 = *(const uint4*)(kp + 24);
      *(uint4*)&Ks[key][dh]      = k0;
      *(uint4*)&Ks[key][dh + 8]  = k1;
      *(uint4*)&Ks[key][dh + 16] = k2;
      *(uint4*)&Ks[key][dh + 24] = k3;
#endif
      const u16* vp = vbase + (size_t)(kb * 128 + key) * QKV_LD + dh;
      uint4 v0 = *(const uint4*)(vp);
      uint4 v1 = *(const uint4*)(vp + 8);
      uint4 v2 = *(const uint4*)(vp + 16);
      uint4 v3 = *(const uint4*)(vp + 24);
      u16 __align__(16) tmp[32];
      *(uint4*)&tmp[0]  = v0;
      *(uint4*)&tmp[8]  = v1;
      *(uint4*)&tmp[16] = v2;
      *(uint4*)&tmp[24] = v3;
#pragma unroll
      for (int i = 0; i < 32; ++i) Vt[dh + i][key] = tmp[i];  // transpose
#if USE_ASYNC_LDS
      __builtin_amdgcn_s_wait_asynccnt(0);
#endif
    }
    __syncthreads();

    // --- scores: S = Q K^T, 8 n-tiles of 16 keys each ---
    float sc[8][8];
#pragma unroll
    for (int t = 0; t < 8; ++t) {
      const int krow = t * 16 + lrow;
      Frag b0, b1;
      b0.u[0] = *(const uint4*)&Ks[krow][lhi * 8];
      b0.u[1] = *(const uint4*)&Ks[krow][16 + lhi * 8];
      b1.u[0] = *(const uint4*)&Ks[krow][32 + lhi * 8];
      b1.u[1] = *(const uint4*)&Ks[krow][48 + lhi * 8];
      v8f s = {};
      s = wmma_bf16(qf[0], b0, s);
      s = wmma_bf16(qf[1], b1, s);
      const bool valid = mrow[kb * 128 + krow] != 0;
#pragma unroll
      for (int r = 0; r < 8; ++r) sc[t][r] = valid ? s[r] : -1.0e10f;
    }

    // --- online softmax (per row; lanes 0-15 = rows r, 16-31 = rows r+8) ---
#pragma unroll
    for (int r = 0; r < 8; ++r) {
      float mx = -3.0e38f;
#pragma unroll
      for (int t = 0; t < 8; ++t) mx = fmaxf(mx, sc[t][r]);
#pragma unroll
      for (int off = 1; off < 16; off <<= 1)
        mx = fmaxf(mx, __shfl_xor(mx, off, 32));
      const float mnew  = fmaxf(mrun[r], mx);
      const float alpha = __expf(mrun[r] - mnew);
      float rs = 0.0f;
#pragma unroll
      for (int t = 0; t < 8; ++t) {
        const float p = __expf(sc[t][r] - mnew);
        sc[t][r] = p;
        rs += p;
      }
#pragma unroll
      for (int off = 1; off < 16; off <<= 1) rs += __shfl_xor(rs, off, 32);
      lrun[r] = lrun[r] * alpha + rs;
      mrun[r] = mnew;
#pragma unroll
      for (int dt = 0; dt < 4; ++dt) o[dt][r] *= alpha;
      // spill P (bf16) to the per-wave LDS tile in row-major
      const int prow = r + lhi * 8;
#pragma unroll
      for (int t = 0; t < 8; ++t)
        Ps[wave][prow][t * 16 + lrow] = f32_to_bf16(sc[t][r]);
    }

    // --- O += P @ V : 4 K-slabs of 32 keys, 4 d-tiles of 16 ---
#pragma unroll
    for (int s4 = 0; s4 < 4; ++s4) {
      Frag pf;
      pf.u[0] = *(const uint4*)&Ps[wave][lrow][s4 * 32 + lhi * 8];
      pf.u[1] = *(const uint4*)&Ps[wave][lrow][s4 * 32 + 16 + lhi * 8];
#pragma unroll
      for (int dt = 0; dt < 4; ++dt) {
        Frag vf;
        const int dcol = dt * 16 + lrow;
        vf.u[0] = *(const uint4*)&Vt[dcol][s4 * 32 + lhi * 8];
        vf.u[1] = *(const uint4*)&Vt[dcol][s4 * 32 + 16 + lhi * 8];
        o[dt] = wmma_bf16(pf, vf, o[dt]);
      }
    }
  }

  // normalize and store (b, s, h*64+d) bf16
#pragma unroll
  for (int r = 0; r < 8; ++r) {
    const float inv = lrun[r] > 0.0f ? 1.0f / lrun[r] : 0.0f;
    const int row = qb * 128 + wave * 16 + r + lhi * 8;
#pragma unroll
    for (int dt = 0; dt < 4; ++dt) {
      const int col = h * 64 + dt * 16 + lrow;
      outb[((size_t)bb * S_LEN + row) * 1024 + col] = f32_to_bf16(o[dt][r] * inv);
    }
  }
}

// ---------------------------------------------------------------------------
// Host-side launcher
// ---------------------------------------------------------------------------
extern "C" void kernel_launch(void* const* d_in, const int* in_sizes, int n_in,
                              void* d_out, int out_size, void* d_ws, size_t ws_size,
                              hipStream_t stream) {
  (void)in_sizes; (void)n_in; (void)out_size; (void)ws_size;

  const float*         x     = (const float*)d_in[0];
  const unsigned char* pm    = (const unsigned char*)d_in[1];  // jnp bool -> 1 byte
  const float*         w_qkv = (const float*)d_in[2];
  const float*         w_o   = (const float*)d_in[3];
  const float*         b_o   = (const float*)d_in[4];
  float*               out   = (float*)d_out;

  // workspace layout (bf16 buffers), 88 MiB total
  char* ws = (char*)d_ws;
  u16* xb    = (u16*)(ws);                                //  8192*1024 (16 MiB)
  u16* wqkvb = (u16*)(ws + (size_t)16 * 1024 * 1024);     //  3072*1024 ( 6 MiB)
  u16* wob   = (u16*)(ws + (size_t)22 * 1024 * 1024);     //  1024*1024 ( 2 MiB)
  u16* qkvb  = (u16*)(ws + (size_t)24 * 1024 * 1024);     //  8192*3072 (48 MiB)
  u16* attnb = (u16*)(ws + (size_t)72 * 1024 * 1024);     //  8192*1024 (16 MiB)

  const float SCALE = 0.125f;  // 64^-0.5, folded into q-rows of w_qkv

  // conversions
  cvt_f32_bf16_x4<<<(8192 * 1024 / 4) / 256, 256, 0, stream>>>(
      (const float4*)x, xb, 8192 * 1024 / 4, 0, 1.0f);
  cvt_f32_bf16_x4<<<(3072 * 1024 / 4) / 256, 256, 0, stream>>>(
      (const float4*)w_qkv, wqkvb, 3072 * 1024 / 4, 1024 * 1024, SCALE);
  cvt_f32_bf16_x4<<<(1024 * 1024 / 4) / 256, 256, 0, stream>>>(
      (const float4*)w_o, wob, 1024 * 1024 / 4, 0, 1.0f);

  // qkv = x @ w_qkv^T  (bf16 out)
  gemm_bf16_nt<false, false><<<dim3(64, 24), 256, 0, stream>>>(
      xb, wqkvb, (void*)qkvb, nullptr, 8192, 3072, 1024);

  // flash attention
  attn_flash<<<1024, 256, 0, stream>>>(qkvb, pm, attnb);

  // out = attn @ w_o^T + b_o  (f32 out)
  gemm_bf16_nt<true, true><<<dim3(64, 8), 256, 0, stream>>>(
      attnb, wob, (void*)out, b_o, 8192, 1024, 1024);
}
